// SparseAutoencoder_7267084665346
// MI455X (gfx1250) — matrix operand order repaired and checked by reading, not compiled, not run on hardware
//
#include <hip/hip_runtime.h>

#define BATCH  16384
#define D_INP  1024
#define D_HID  16384
#define TOPK   64

#define MTILE  64          // batch rows per workgroup
#define HTILE  128         // hidden columns per tile
#define KTILE  256         // k-chunk staged in LDS per step
#define NCHUNK ((D_HID / HTILE) * (D_INP / KTILE))   // 512

#define LDSX_OFF   0
#define LDSX_BYTES (MTILE * D_INP * 2)               // 131072: x rows, bf16
#define LDSW_OFF   (LDSX_OFF + LDSX_BYTES)
#define LDSW_BYTES (HTILE * KTILE * 2)               // 65536 per buffer
#define ENC_OFF    (LDSW_OFF + 2 * LDSW_BYTES)       // double-buffered W
#define ENC_BYTES  (MTILE * HTILE * 4)               // 32768: encoded tile
#define SMEM_BYTES (ENC_OFF + ENC_BYTES)             // 294912 <= 320KB

typedef __attribute__((ext_vector_type(16))) __bf16        v16bf;
typedef __attribute__((ext_vector_type(8)))  float         v8f;
typedef __attribute__((ext_vector_type(4)))  float         v4f;
typedef __attribute__((ext_vector_type(4)))  unsigned int  v4u;
typedef __attribute__((ext_vector_type(2)))  unsigned int  v2u;

union FragAB { v4u q[2]; v16bf v; };

__device__ __forceinline__ unsigned short bf16_rne(float f) {
  unsigned int u = __float_as_uint(f);
  u += 0x7fffu + ((u >> 16) & 1u);               // round-to-nearest-even
  return (unsigned short)(u >> 16);
}

// ---------------- kernel 1: f32 -> bf16 repack into workspace ---------------
__global__ __launch_bounds__(256) void cvt_bf16_kernel(const float* __restrict__ src,
                                                       unsigned short* __restrict__ dst,
                                                       int n4) {
  int t = blockIdx.x * 256 + threadIdx.x;
  int stride = gridDim.x * 256;
  for (int i = t; i < n4; i += stride) {
    v4f a = ((const v4f*)src)[i];
    v2u o;
    o.x = (unsigned)bf16_rne(a.x) | ((unsigned)bf16_rne(a.y) << 16);
    o.y = (unsigned)bf16_rne(a.z) | ((unsigned)bf16_rne(a.w) << 16);
    ((v2u*)dst)[i] = o;
  }
}

// ---------------- CDNA5 async copy (ASYNCcnt path) --------------------------
__device__ __forceinline__ void async_b128(unsigned int lds_off, const void* base,
                                           unsigned int voff) {
  asm volatile("global_load_async_to_lds_b128 %0, %1, %2"
               :: "v"(lds_off), "v"(voff), "s"(base) : "memory");
}
__device__ __forceinline__ void wait_async0()  {
  asm volatile("s_wait_asynccnt 0" ::: "memory");
}
__device__ __forceinline__ void wait_async16() {
  asm volatile("s_wait_asynccnt 16" ::: "memory");
}

// issue the 16 per-thread async b128 loads staging W chunk `chunk` (hc,ks)
__device__ __forceinline__ void issue_wchunk(const unsigned short* wbf, int chunk,
                                             unsigned tid) {
  const int hc = chunk >> 2;
  const int ks = chunk & 3;
  const unsigned h0 = (unsigned)hc * HTILE;
  const unsigned k0 = (unsigned)ks * KTILE;
  const unsigned base = LDSW_OFF + (unsigned)(chunk & 1) * LDSW_BYTES;
#pragma unroll 1
  for (unsigned i = tid; i < LDSW_BYTES / 16; i += 256) {
    unsigned r = i >> 5;                         // 0..127, 32 x 16B per row
    unsigned c = i & 31;
    async_b128(base + i * 16, wbf,
               (h0 + r) * (D_INP * 2) + k0 * 2 + c * 16);
  }
}

// ---------------- kernel 2: fused encode(WMMA bf16) + top-k + decode --------
__global__ __launch_bounds__(256) void sae_fused_kernel(
    const unsigned short* __restrict__ xbf,   // [BATCH][D_INP] bf16
    const unsigned short* __restrict__ wbf,   // [D_HID][D_INP] bf16
    const float* __restrict__ Wenc,           // [D_HID][D_INP] f32 (decode)
    const float* __restrict__ bdec,           // [D_INP] f32
    float* __restrict__ out)                  // [BATCH][D_INP] f32
{
  extern __shared__ unsigned char smem[];

  const int lane  = threadIdx.x & 31;
  const int wave  = threadIdx.x >> 5;         // 8 waves
  const int khalf = lane >> 4;
  const int l15   = lane & 15;
  const int row0  = blockIdx.x * MTILE;

  // ---- stage 64 x-rows (bf16) into LDS + prefetch first W chunk ----
  for (int i = threadIdx.x; i < LDSX_BYTES / 16; i += 256) {
    async_b128((unsigned)(LDSX_OFF + i * 16), xbf,
               (unsigned)row0 * (D_INP * 2) + (unsigned)i * 16);
  }
  issue_wchunk(wbf, 0, threadIdx.x);
  wait_async16();                              // x rows landed (in-order)
  __syncthreads();

  // ---- running top-64 per row: 2 entries per lane, 8 rows per wave ----
  float tv0[8], tv1[8];
  unsigned int ti0[8], ti1[8];
  float mval[8];
  int   mpos[8];
#pragma unroll
  for (int r = 0; r < 8; ++r) {
    tv0[r] = -INFINITY; tv1[r] = -INFINITY;
    ti0[r] = 0u;        ti1[r] = 0u;
    mval[r] = -INFINITY; mpos[r] = 0;
  }

  for (int hc = 0; hc < D_HID / HTILE; ++hc) {     // 128 column tiles
    const int h0 = hc * HTILE;
    v8f acc0 = {0,0,0,0,0,0,0,0};
    v8f acc1 = {0,0,0,0,0,0,0,0};
    v8f acc2 = {0,0,0,0,0,0,0,0};
    v8f acc3 = {0,0,0,0,0,0,0,0};

    for (int ks = 0; ks < D_INP / KTILE; ++ks) {   // 4 k-chunks of 256
      const int chunk = hc * 4 + ks;
      const int k0 = ks * KTILE;
      const unsigned wbase = LDSW_OFF + (unsigned)(chunk & 1) * LDSW_BYTES;

      __syncthreads();                 // buf[(chunk+1)&1] readers (chunk-1) done
      if (chunk + 1 < NCHUNK) {
        issue_wchunk(wbf, chunk + 1, threadIdx.x);   // pipeline next chunk
        wait_async16();                // current chunk landed (in-order)
      } else {
        wait_async0();
      }
      __syncthreads();                 // all waves' transfers visible

#pragma unroll
      for (int kk = 0; kk < KTILE / 32; ++kk) {    // 8 WMMA k-steps
        const int kb = kk * 32;
        // B fragment: lane = N (h col), lanes0-15 K=0..7,16..23, lanes16-31 K=8..15,24..31
        FragAB b;
        const unsigned bb = wbase + (unsigned)(wave * 16 + l15) * (KTILE * 2)
                          + (unsigned)(kb + khalf * 8) * 2;
        b.q[0] = *(const v4u*)(smem + bb);
        b.q[1] = *(const v4u*)(smem + bb + 32);
        // A fragments (all four loaded up front -> one dscnt wait per group)
        FragAB a0, a1, a2, a3;
        const unsigned ak = (unsigned)(k0 + kb + khalf * 8) * 2;
        const unsigned ar = (unsigned)l15 * (D_INP * 2);
        a0.q[0] = *(const v4u*)(smem + (ar + ak));
        a0.q[1] = *(const v4u*)(smem + (ar + ak) + 32);
        a1.q[0] = *(const v4u*)(smem + (ar + 16 * (D_INP * 2) + ak));
        a1.q[1] = *(const v4u*)(smem + (ar + 16 * (D_INP * 2) + ak) + 32);
        a2.q[0] = *(const v4u*)(smem + (ar + 32 * (D_INP * 2) + ak));
        a2.q[1] = *(const v4u*)(smem + (ar + 32 * (D_INP * 2) + ak) + 32);
        a3.q[0] = *(const v4u*)(smem + (ar + 48 * (D_INP * 2) + ak));
        a3.q[1] = *(const v4u*)(smem + (ar + 48 * (D_INP * 2) + ak) + 32);
        acc0 = __builtin_amdgcn_wmma_f32_16x16x32_bf16(
                   false, a0.v, false, b.v, (short)0, acc0, false, false);
        acc1 = __builtin_amdgcn_wmma_f32_16x16x32_bf16(
                   false, a1.v, false, b.v, (short)0, acc1, false, false);
        acc2 = __builtin_amdgcn_wmma_f32_16x16x32_bf16(
                   false, a2.v, false, b.v, (short)0, acc2, false, false);
        acc3 = __builtin_amdgcn_wmma_f32_16x16x32_bf16(
                   false, a3.v, false, b.v, (short)0, acc3, false, false);
      }
    }

    // ---- spill encoded 64x128 tile to LDS (C layout: lane half = M+8) ----
#pragma unroll
    for (int r = 0; r < 8; ++r) {
      const unsigned col = (unsigned)(wave * 16 + l15) * 4;
      *(float*)(smem + ENC_OFF + (unsigned)( 0 + r + 8 * khalf) * (HTILE * 4) + col) = acc0[r];
      *(float*)(smem + ENC_OFF + (unsigned)(16 + r + 8 * khalf) * (HTILE * 4) + col) = acc1[r];
      *(float*)(smem + ENC_OFF + (unsigned)(32 + r + 8 * khalf) * (HTILE * 4) + col) = acc2[r];
      *(float*)(smem + ENC_OFF + (unsigned)(48 + r + 8 * khalf) * (HTILE * 4) + col) = acc3[r];
    }
    __syncthreads();

    // ---- streaming top-64 update, one wave per 8 rows ----
#pragma unroll
    for (int rr = 0; rr < 8; ++rr) {
      const int row = wave * 8 + rr;
#pragma unroll
      for (int seg = 0; seg < HTILE / 32; ++seg) {
        const int col = seg * 32 + lane;
        float v = *(const float*)(smem + ENC_OFF + (unsigned)row * (HTILE * 4)
                                  + (unsigned)col * 4);
        unsigned int idx = (unsigned)(h0 + col);
        unsigned int mask = (unsigned int)__ballot(v > mval[rr]);
        while (mask) {
          int src = __builtin_ctz(mask);
          mask &= mask - 1;
          float cv = __shfl(v, src);
          unsigned int ci = __shfl(idx, src);
          if (cv > mval[rr]) {
            int p = mpos[rr];
            if (p < 32) { if (lane == p)      { tv0[rr] = cv; ti0[rr] = ci; } }
            else        { if (lane == p - 32) { tv1[rr] = cv; ti1[rr] = ci; } }
            // wave-wide argmin over the 64-entry list (butterfly)
            float lv = tv0[rr]; int lp = lane;
            if (tv1[rr] < lv) { lv = tv1[rr]; lp = lane + 32; }
#pragma unroll
            for (int off = 16; off; off >>= 1) {
              float ov = __shfl_xor(lv, off);
              int   op = __shfl_xor(lp, off);
              if (ov < lv || (ov == lv && op < lp)) { lv = ov; lp = op; }
            }
            mval[rr] = lv; mpos[rr] = lp;
          }
        }
      }
    }
    __syncthreads();
  }

  // ---- sparse decode: out[row] = sum_j val_j * Wenc[idx_j, :] + b_dec ----
#pragma unroll 1
  for (int rr = 0; rr < 8; ++rr) {
    const int row = wave * 8 + rr;
    v4f racc[8];
#pragma unroll
    for (int c = 0; c < 8; ++c)
      racc[c] = *(const v4f*)(bdec + c * 128 + lane * 4);
#pragma unroll 1
    for (int slot = 0; slot < 2; ++slot) {
#pragma unroll 1
      for (int jj = 0; jj < 32; ++jj) {
        float        cv = __shfl(slot == 0 ? tv0[rr] : tv1[rr], jj);
        unsigned int ci = __shfl(slot == 0 ? ti0[rr] : ti1[rr], jj);
        const v4f* wrow = (const v4f*)(Wenc + (size_t)ci * D_INP);
#pragma unroll
        for (int c = 0; c < 8; ++c) {
          v4f wv = wrow[c * 32 + lane];
          racc[c].x += cv * wv.x; racc[c].y += cv * wv.y;
          racc[c].z += cv * wv.z; racc[c].w += cv * wv.w;
        }
      }
    }
    float* orow = out + (size_t)(row0 + row) * D_INP;
#pragma unroll
    for (int c = 0; c < 8; ++c)
      *(v4f*)(orow + c * 128 + lane * 4) = racc[c];
  }
}

// ---------------------------------------------------------------------------
extern "C" void kernel_launch(void* const* d_in, const int* in_sizes, int n_in,
                              void* d_out, int out_size, void* d_ws, size_t ws_size,
                              hipStream_t stream) {
  const float* x    = (const float*)d_in[0];   // [16384,1024]
  const float* Wenc = (const float*)d_in[1];   // [16384,1024]
  const float* bdec = (const float*)d_in[2];   // [1024]
  float* out = (float*)d_out;

  unsigned short* xbf = (unsigned short*)d_ws;                           // 32MB
  unsigned short* wbf = (unsigned short*)((char*)d_ws +
                        (size_t)BATCH * D_INP * sizeof(unsigned short)); // 32MB

  const int n4 = (BATCH * D_INP) / 4;          // both tensors: 16M elems each
  cvt_bf16_kernel<<<4096, 256, 0, stream>>>(x,    xbf, n4);
  cvt_bf16_kernel<<<4096, 256, 0, stream>>>(Wenc, wbf, n4);

  (void)hipFuncSetAttribute((const void*)sae_fused_kernel,
                            hipFuncAttributeMaxDynamicSharedMemorySize,
                            SMEM_BYTES);
  sae_fused_kernel<<<BATCH / MTILE, 256, SMEM_BYTES, stream>>>(
      xbf, wbf, Wenc, bdec, out);
}
